// Spatio_Temporal_Att_27419071217795
// MI455X (gfx1250) — matrix operand order repaired and checked by reading, not compiled, not run on hardware
//
#include <hip/hip_runtime.h>
#include <math.h>

#define HH   4
#define CC   32
#define HID  128
#define BB   4
#define TT   16
#define NNODE 500
#define FF   60
#define EE   4000
#define UU   50
#define LBL  10
#define MROWS (BB*TT*NNODE)   // 32000

typedef __attribute__((ext_vector_type(16))) __bf16 v16bf;
typedef __attribute__((ext_vector_type(8)))  float  v8f;

__device__ __forceinline__ unsigned short f2bf(float f) {
  unsigned int u = __float_as_uint(f);
  unsigned int r = 0x7FFFu + ((u >> 16) & 1u);
  return (unsigned short)((u + r) >> 16);
}

__device__ __forceinline__ void atomicMaxF(float* addr, float val) {
  if (val >= 0.0f) atomicMax((int*)addr, __float_as_int(val));
  else             atomicMin((unsigned int*)addr, __float_as_uint(val));
}

// ---------------- utility kernels ----------------
__global__ void fill_kernel(float* p, float v, int n) {
  int i = blockIdx.x * blockDim.x + threadIdx.x;
  if (i < n) p[i] = v;
}

__global__ void w2bf_kernel(const float* __restrict__ w, unsigned short* __restrict__ wb, int n) {
  int i = blockIdx.x * blockDim.x + threadIdx.x;
  if (i < n) wb[i] = f2bf(w[i]);
}

// x[B,T,N,F] + sinusoid PE(t) -> XB bf16 [M,64]
__global__ void concat_pe_kernel(const float* __restrict__ x, unsigned short* __restrict__ xb) {
  int idx = blockIdx.x * blockDim.x + threadIdx.x;
  if (idx >= MROWS * 64) return;
  int k = idx & 63;
  int row = idx >> 6;
  float val;
  if (k < FF) {
    val = x[(size_t)row * FF + k];
  } else {
    int s = row / NNODE;
    float tf = (float)(s % TT);
    int kk = k - FF;                    // pe = {sin(t), cos(t), sin(.01t), cos(.01t)}
    val = (kk == 0) ? sinf(tf) : (kk == 1) ? cosf(tf)
        : (kk == 2) ? sinf(0.01f * tf) : cosf(0.01f * tf);
  }
  xb[(size_t)row * 64 + k] = f2bf(val);
}

// ---------------- WMMA GEMM: C[M,128] = A[M,K](bf16) * W[128,K](bf16)^T + bias ----------------
// Register-blocked: one wave32 owns a 16-row block and all 8 column tiles
// (8 x v8f accumulators). A fragment loaded once per K-step, reused by 8 WMMAs.
template<int K>
__global__ void gemm_bf16_wmma(const unsigned short* __restrict__ A,
                               const unsigned short* __restrict__ W,
                               const float* __restrict__ bias,
                               float* __restrict__ C, int M) {
  int wv   = blockIdx.x * (blockDim.x >> 5) + (threadIdx.x >> 5);
  int lane = threadIdx.x & 31;
  if (wv >= (M >> 4)) return;
  int h = lane >> 4, r = lane & 15;

  const unsigned short* arow = A + (size_t)(wv * 16 + r) * K;

  v8f acc[8];
#pragma unroll
  for (int t = 0; t < 8; ++t) acc[t] = (v8f){};

#pragma unroll
  for (int kb = 0; kb < K; kb += 32) {
    union { uint4 u[2]; v16bf v; } af;
    af.u[0] = *(const uint4*)(arow + kb + h * 8);        // A lane: K = kb+8h .. +8
    af.u[1] = *(const uint4*)(arow + kb + 16 + h * 8);   //         K = kb+16+8h .. +8
    if (kb + 32 < K) __builtin_prefetch((const void*)(arow + kb + 32), 0, 1);
#pragma unroll
    for (int tn = 0; tn < 8; ++tn) {
      const unsigned short* brow = W + (size_t)(tn * 16 + r) * K;
      union { uint4 u[2]; v16bf v; } bf;
      bf.u[0] = *(const uint4*)(brow + kb + h * 8);      // B lane: K = kb+8h .. +8
      bf.u[1] = *(const uint4*)(brow + kb + 16 + h * 8); //         K = kb+16+8h .. +8
      acc[tn] = __builtin_amdgcn_wmma_f32_16x16x32_bf16(false, af.v, false, bf.v,
                                                        (short)0, acc[tn], false, false);
    }
  }
#pragma unroll
  for (int tn = 0; tn < 8; ++tn) {
    int col = tn * 16 + r;
    float b = bias[col];
#pragma unroll
    for (int rr = 0; rr < 8; ++rr) {
      int row = wv * 16 + rr + 8 * h;
      C[(size_t)row * HID + col] = acc[tn][rr] + b;
    }
  }
}

// ---------------- edge attention (generic over spatial / dense-temporal) ----------------
__global__ void logits_kernel(const float* __restrict__ Q, const float* __restrict__ Kf,
                              const int* __restrict__ srcIdx, const int* __restrict__ dstIdx,
                              int denseT, int Ecnt, int Scnt, int ncnt,
                              float* __restrict__ Alog, float* __restrict__ MX) {
  int idx = blockIdx.x * blockDim.x + threadIdx.x;
  if (idx >= Ecnt * Scnt * HH) return;
  int h = idx % HH; int rem = idx / HH;
  int s = rem % Scnt; int e = rem / Scnt;
  int src, dst;
  if (denseT > 0) { src = e / denseT; dst = e % denseT; }
  else            { src = srcIdx[e];  dst = dstIdx[e]; }
  const float* q = Q  + (size_t)(s * ncnt + dst) * HID + h * CC;
  const float* k = Kf + (size_t)(s * ncnt + src) * HID + h * CC;
  float a = 0.0f;
#pragma unroll
  for (int c = 0; c < CC; ++c) a += q[c] * k[c];
  a *= 0.17677669529663687f;   // 1/sqrt(32)
  Alog[idx] = a;
  atomicMaxF(MX + (size_t)(dst * Scnt + s) * HH + h, a);
}

__global__ void expden_kernel(float* __restrict__ Alog, const float* __restrict__ MX,
                              float* __restrict__ DEN,
                              const int* __restrict__ dstIdx, int denseT,
                              int Ecnt, int Scnt) {
  int idx = blockIdx.x * blockDim.x + threadIdx.x;
  if (idx >= Ecnt * Scnt * HH) return;
  int h = idx % HH; int rem = idx / HH;
  int s = rem % Scnt; int e = rem / Scnt;
  int dst = (denseT > 0) ? (e % denseT) : dstIdx[e];
  size_t seg = (size_t)(dst * Scnt + s) * HH + h;
  float ex = __expf(Alog[idx] - MX[seg]);
  Alog[idx] = ex;
  atomicAdd(DEN + seg, ex);
}

__global__ void msg_kernel(const float* __restrict__ V, const float* __restrict__ Alog,
                           const float* __restrict__ DEN, float* __restrict__ AGG,
                           const int* __restrict__ srcIdx, const int* __restrict__ dstIdx,
                           int denseT, int Ecnt, int Scnt, int ncnt) {
  int idx = blockIdx.x * blockDim.x + threadIdx.x;
  if (idx >= Ecnt * Scnt * HH) return;
  int h = idx % HH; int rem = idx / HH;
  int s = rem % Scnt; int e = rem / Scnt;
  int src, dst;
  if (denseT > 0) { src = e / denseT; dst = e % denseT; }
  else            { src = srcIdx[e];  dst = dstIdx[e]; }
  float w = Alog[idx] / DEN[(size_t)(dst * Scnt + s) * HH + h];
  const float* v = V + (size_t)(s * ncnt + src) * HID + h * CC;
  float* ag      = AGG + (size_t)(s * ncnt + dst) * HID + h * CC;
#pragma unroll
  for (int c = 0; c < CC; ++c) atomicAdd(ag + c, v[c] * w);
}

// skip-add + LayerNorm + ELU, one wave32 per 128-wide row; writes f32 + bf16
__global__ void ln_elu_kernel(const float* __restrict__ agg, const float* __restrict__ skip,
                              const float* __restrict__ g, const float* __restrict__ bta,
                              float* __restrict__ xout, unsigned short* __restrict__ xbout,
                              int Mrows) {
  int wv   = blockIdx.x * (blockDim.x >> 5) + (threadIdx.x >> 5);
  int lane = threadIdx.x & 31;
  if (wv >= Mrows) return;
  const float* pa = agg  + (size_t)wv * HID;
  const float* ps = skip + (size_t)wv * HID;
  float v[4];
  float sum = 0.0f;
#pragma unroll
  for (int i = 0; i < 4; ++i) { v[i] = pa[lane * 4 + i] + ps[lane * 4 + i]; sum += v[i]; }
#pragma unroll
  for (int o = 16; o; o >>= 1) sum += __shfl_xor(sum, o, 32);
  float mu = sum * (1.0f / HID);
  float var = 0.0f;
#pragma unroll
  for (int i = 0; i < 4; ++i) { float d = v[i] - mu; var += d * d; }
#pragma unroll
  for (int o = 16; o; o >>= 1) var += __shfl_xor(var, o, 32);
  float inv = rsqrtf(var * (1.0f / HID) + 1e-5f);
#pragma unroll
  for (int i = 0; i < 4; ++i) {
    int c = lane * 4 + i;
    float y = (v[i] - mu) * inv * g[c] + bta[c];
    y = (y > 0.0f) ? y : (__expf(y) - 1.0f);          // ELU
    xout[(size_t)wv * HID + c]  = y;
    xbout[(size_t)wv * HID + c] = f2bf(y);
  }
}

// X f32 [B,T,N,128] -> XB bf16 [B,N,T,128]
__global__ void transpose_kernel(const float* __restrict__ X, unsigned short* __restrict__ XB) {
  int idx = blockIdx.x * blockDim.x + threadIdx.x;
  if (idx >= MROWS * HID) return;
  int c = idx & (HID - 1);
  int r = idx >> 7;
  int t = r % TT; int r2 = r / TT;
  int nn = r2 % NNODE; int b = r2 / NNODE;
  XB[idx] = f2bf(X[((size_t)((b * TT + t) * NNODE + nn)) * HID + c]);
}

// mean over T: X [B,N,T,128] -> XM [B,N,128]
__global__ void tpool_kernel(const float* __restrict__ X, float* __restrict__ XM) {
  int idx = blockIdx.x * blockDim.x + threadIdx.x;
  if (idx >= BB * NNODE * HID) return;
  int c = idx & (HID - 1);
  int r = idx >> 7;
  float s = 0.0f;
#pragma unroll
  for (int t = 0; t < TT; ++t) s += X[((size_t)r * TT + t) * HID + c];
  XM[idx] = s * (1.0f / TT);
}

__global__ void cnt_kernel(const int* __restrict__ unit, float* __restrict__ CNT) {
  int i = blockIdx.x * blockDim.x + threadIdx.x;
  if (i < NNODE) atomicAdd(CNT + unit[i], 1.0f);
}

__global__ void usum_kernel(const float* __restrict__ XM, const int* __restrict__ unit,
                            float* __restrict__ XU) {
  int idx = blockIdx.x * blockDim.x + threadIdx.x;
  if (idx >= BB * NNODE * HID) return;
  int c = idx & (HID - 1);
  int r = idx >> 7;
  int nn = r % NNODE; int b = r / NNODE;
  atomicAdd(&XU[((size_t)(b * UU + unit[nn])) * HID + c], XM[idx]);
}

__global__ void udiv_kernel(float* __restrict__ XU, const float* __restrict__ CNT) {
  int idx = blockIdx.x * blockDim.x + threadIdx.x;
  if (idx >= BB * UU * HID) return;
  int u = (idx >> 7) % UU;
  XU[idx] /= fmaxf(CNT[u], 1.0f);
}

__global__ void task_kernel(const float* __restrict__ XU, const float* __restrict__ tW,
                            const float* __restrict__ tB, float* __restrict__ out) {
  int idx = blockIdx.x * blockDim.x + threadIdx.x;
  if (idx >= BB * UU) return;
  const float* xu = XU + (size_t)idx * HID;
  float lg[LBL]; float mx = -3.0e38f;
  for (int l = 0; l < LBL; ++l) {
    float s = tB[l];
    for (int c = 0; c < HID; ++c) s += xu[c] * tW[l * HID + c];
    lg[l] = s; mx = fmaxf(mx, s);
  }
  float den = 0.0f;
  for (int l = 0; l < LBL; ++l) { lg[l] = __expf(lg[l] - mx); den += lg[l]; }
  for (int l = 0; l < LBL; ++l) out[(size_t)idx * LBL + l] = lg[l] / den;
}

__global__ void atk_kernel(const float* __restrict__ XU, const float* __restrict__ aW,
                           const float* __restrict__ aB, float* __restrict__ out) {
  int idx = blockIdx.x * blockDim.x + threadIdx.x;
  if (idx >= BB * UU * UU) return;
  int vv = idx % UU; int r = idx / UU;
  int u = r % UU; int b = r / UU;
  const float* pu = XU + ((size_t)(b * UU + u))  * HID;
  const float* pv = XU + ((size_t)(b * UU + vv)) * HID;
  float s = aB[0];
  for (int c = 0; c < HID; ++c) s += pu[c] * pv[c] * aW[c];
  out[BB * UU * LBL + idx] = 1.0f / (1.0f + __expf(-s));
}

#define GRID(n, b) (((n) + (b) - 1) / (b))

extern "C" void kernel_launch(void* const* d_in, const int* in_sizes, int n_in,
                              void* d_out, int out_size, void* d_ws, size_t ws_size,
                              hipStream_t stream) {
  (void)in_sizes; (void)n_in; (void)out_size; (void)ws_size;
  const float* x     = (const float*)d_in[0];
  const int*   edge  = (const int*)d_in[1];
  const int*   unit  = (const int*)d_in[2];
  const float* atkW  = (const float*)d_in[3];
  const float* atkB  = (const float*)d_in[4];
  const float* taskW = (const float*)d_in[45];
  const float* taskB = (const float*)d_in[46];
  float* out = (float*)d_out;

  char* ws = (char*)d_ws;
  size_t off = 0;
  auto alloc = [&](size_t bytes) -> void* {
    void* p = ws + off;
    off = (off + bytes + 255) & ~(size_t)255;
    return p;
  };
  unsigned short* XB = (unsigned short*)alloc((size_t)MROWS * HID * 2);
  unsigned short* WB = (unsigned short*)alloc((size_t)HID * HID * 2);
  float* Q    = (float*)alloc((size_t)MROWS * HID * 4);
  float* Kb   = (float*)alloc((size_t)MROWS * HID * 4);
  float* V    = (float*)alloc((size_t)MROWS * HID * 4);
  float* SK   = (float*)alloc((size_t)MROWS * HID * 4);
  float* ALOG = (float*)alloc((size_t)2048000 * 4);         // max E*S*H
  float* MX   = (float*)alloc((size_t)128000 * 4);          // n*S*H (same both stages)
  float* DEN  = (float*)alloc((size_t)128000 * 4);
  float* X    = (float*)alloc((size_t)MROWS * HID * 4);
  float* XM   = (float*)alloc((size_t)BB * NNODE * HID * 4);
  float* XU   = (float*)alloc((size_t)BB * UU * HID * 4);
  float* CNT  = (float*)alloc((size_t)UU * 4);
  float* AGG  = Q;   // Q is dead after logits_kernel; reuse for aggregation

  concat_pe_kernel<<<GRID(MROWS * 64, 256), 256, 0, stream>>>(x, XB);

  int din = 64;
  for (int layer = 0; layer < 4; ++layer) {
    int base = 5 + layer * 10;   // JAX pytree order: k.W,k.b,ln_b,ln_g,q.W,q.b,s.W,s.b,v.W,v.b
    const float* kW  = (const float*)d_in[base + 0];
    const float* kB_ = (const float*)d_in[base + 1];
    const float* lnB = (const float*)d_in[base + 2];
    const float* lnG = (const float*)d_in[base + 3];
    const float* qW  = (const float*)d_in[base + 4];
    const float* qB  = (const float*)d_in[base + 5];
    const float* sW  = (const float*)d_in[base + 6];
    const float* sB  = (const float*)d_in[base + 7];
    const float* vW  = (const float*)d_in[base + 8];
    const float* vB  = (const float*)d_in[base + 9];

    const float* Ws[4] = {qW, kW, vW, sW};
    const float* Bs[4] = {qB, kB_, vB, sB};
    float* Outs[4] = {Q, Kb, V, SK};
    int waves = MROWS / 16;                 // one wave per 16-row block (all 128 cols)
    for (int p = 0; p < 4; ++p) {
      w2bf_kernel<<<GRID(HID * din, 256), 256, 0, stream>>>(Ws[p], WB, HID * din);
      if (din == 64)
        gemm_bf16_wmma<64><<<GRID(waves, 8), 256, 0, stream>>>(XB, WB, Bs[p], Outs[p], MROWS);
      else
        gemm_bf16_wmma<128><<<GRID(waves, 8), 256, 0, stream>>>(XB, WB, Bs[p], Outs[p], MROWS);
    }

    bool spatial = (layer < 2);
    int Ecnt   = spatial ? EE : TT * TT;
    int Scnt   = spatial ? BB * TT : BB * NNODE;
    int ncnt   = spatial ? NNODE : TT;
    int denseT = spatial ? 0 : TT;
    const int* srcIdx = spatial ? edge : nullptr;
    const int* dstIdx = spatial ? (edge + EE) : nullptr;
    int segTot  = ncnt * Scnt * HH;           // 128000 both stages
    int edgeTot = Ecnt * Scnt * HH;

    fill_kernel<<<GRID(segTot, 256), 256, 0, stream>>>(MX, -3.0e38f, segTot);
    fill_kernel<<<GRID(segTot, 256), 256, 0, stream>>>(DEN, 0.0f, segTot);
    logits_kernel<<<GRID(edgeTot, 256), 256, 0, stream>>>(Q, Kb, srcIdx, dstIdx, denseT,
                                                          Ecnt, Scnt, ncnt, ALOG, MX);
    fill_kernel<<<GRID(MROWS * HID, 256), 256, 0, stream>>>(AGG, 0.0f, MROWS * HID);
    expden_kernel<<<GRID(edgeTot, 256), 256, 0, stream>>>(ALOG, MX, DEN, dstIdx, denseT, Ecnt, Scnt);
    msg_kernel<<<GRID(edgeTot, 256), 256, 0, stream>>>(V, ALOG, DEN, AGG, srcIdx, dstIdx,
                                                       denseT, Ecnt, Scnt, ncnt);
    ln_elu_kernel<<<GRID(MROWS, 8), 256, 0, stream>>>(AGG, SK, lnG, lnB, X, XB, MROWS);

    din = HID;
    if (layer == 1)   // [B,T,N,hid] -> [B,N,T,hid] for the temporal stage
      transpose_kernel<<<GRID(MROWS * HID, 256), 256, 0, stream>>>(X, XB);
  }

  tpool_kernel<<<GRID(BB * NNODE * HID, 256), 256, 0, stream>>>(X, XM);
  fill_kernel<<<GRID(BB * UU * HID, 256), 256, 0, stream>>>(XU, 0.0f, BB * UU * HID);
  fill_kernel<<<1, 64, 0, stream>>>(CNT, 0.0f, UU);
  cnt_kernel<<<GRID(NNODE, 64), 64, 0, stream>>>(unit, CNT);
  usum_kernel<<<GRID(BB * NNODE * HID, 256), 256, 0, stream>>>(XM, unit, XU);
  udiv_kernel<<<GRID(BB * UU * HID, 256), 256, 0, stream>>>(XU, CNT);
  task_kernel<<<GRID(BB * UU, 64), 64, 0, stream>>>(XU, taskW, taskB, out);
  atk_kernel<<<GRID(BB * UU * UU, 256), 256, 0, stream>>>(XU, atkW, atkB, out);
}
